// MultiHeadAttention_77403900608651
// MI455X (gfx1250) — matrix operand order repaired and checked by reading, not compiled, not run on hardware
//
#include <hip/hip_runtime.h>
#include <hip/hip_bf16.h>

// ---------------------------------------------------------------------------
// MHA forward for MI455X (gfx1250), bf16 WMMA with f32 accumulation.
//   B=2, S=2048, E=1024, H=16, D=64
// Pipeline:
//   1) f32 -> bf16 conversion (vectorized) of activations and weights
//   2) Register-blocked WMMA GEMM (32x64 tile/wave, 8 accumulators):
//        Q = X Wq^T + bq  -> [b,h,s,d] bf16
//        K = X Wk^T + bk  -> [b,h,s,d] bf16
//        V = X Wv^T + bv  -> [b,h,d,s] bf16 (transposed for P@V)
//   3) Flash attention per 16-row Q tile per wave (online softmax),
//      P-tile transposed C->A layout via LDS + s_wait_dscnt
//   4) WMMA GEMM: out = Ctx Wo^T + bo -> f32 d_out
// ---------------------------------------------------------------------------

#define EMBED 1024
#define HEADS 16
#define HDIM  64
#define BATCH 2
#define SEQ   2048
#define TOKENS (BATCH * SEQ)   // 4096

typedef __bf16 bf16_t;
typedef __attribute__((ext_vector_type(4)))  float  f32x4;
typedef __attribute__((ext_vector_type(4)))  __bf16 bf16x4;
typedef __attribute__((ext_vector_type(8)))  __bf16 bf16x8;
typedef __attribute__((ext_vector_type(16))) __bf16 bf16x16;
typedef __attribute__((ext_vector_type(8)))  float  f32x8;

__device__ __forceinline__ f32x8 wmma_bf16(bf16x16 a, bf16x16 b, f32x8 c) {
  // 8 args: (neg_a, A, neg_b, B, c_mod, C, reuse_a, reuse_b)
  return __builtin_amdgcn_wmma_f32_16x16x32_bf16(
      false, a, false, b, (short)0, c, false, false);
}

__device__ __forceinline__ bf16x16 concat8(bf16x8 c0, bf16x8 c1) {
  return __builtin_shufflevector(c0, c1, 0,1,2,3,4,5,6,7,8,9,10,11,12,13,14,15);
}

// A fragment (16x32 bf16, M x K) from a row-major tile with row stride ldk.
// lanes 0-15: row=lane,    K {0..15}
// lanes16-31: row=lane-16, K {16..31}
__device__ __forceinline__ bf16x16 frag_a(const bf16_t* tile, int ldk, int lane) {
  const int row  = lane & 15;
  const int koff = (lane & 16) ? 16 : 0;
  const bf16_t* p = tile + row * ldk + koff;
  bf16x8 c0 = *(const bf16x8*)(p);
  bf16x8 c1 = *(const bf16x8*)(p + 8);
  return concat8(c0, c1);
}

// B fragment (32x16 bf16, K x N): lane = column n, 16 contiguous K values.
// Caller passes the pointer to this lane's K-run start (already includes the
// +16 element shift for lanes 16-31).
__device__ __forceinline__ bf16x16 frag_b(const bf16_t* p) {
  bf16x8 c0 = *(const bf16x8*)(p);
  bf16x8 c1 = *(const bf16x8*)(p + 8);
  return concat8(c0, c1);
}

// ---------------------------------------------------------------------------
// Vectorized conversion: 4 floats -> 4 bf16 per thread (n multiple of 4).
__global__ void cvt_f32_to_bf16(const float* __restrict__ in,
                                bf16_t* __restrict__ out, int n4) {
  int i = blockIdx.x * blockDim.x + threadIdx.x;
  if (i < n4) {
    f32x4 v = *((const f32x4*)in + i);
    bf16x4 o;
    o[0] = (bf16_t)v[0]; o[1] = (bf16_t)v[1];
    o[2] = (bf16_t)v[2]; o[3] = (bf16_t)v[3];
    *((bf16x4*)out + i) = o;
  }
}

// ---------------------------------------------------------------------------
// Y = A @ W^T + bias, A:[4096,1024] bf16, W:[1024,1024] bf16 (torch Linear
// weight, row n = output feature). One wave -> 32x64 output tile:
// 2 M-fragments x 4 N-fragments, 8 f32x8 accumulators. Per k-slice:
// 4 A-loads + 8 B-loads feed 8 WMMA (1.5 b128 loads per WMMA).
// mode 0: bf16 out, layout [b,h,s,d]
// mode 1: bf16 out, layout [b,h,d,s]   (transposed V)
// mode 2: f32  out, layout [t, e]      (final output)
__global__ __launch_bounds__(128)
void gemm_bf16_wmma(const bf16_t* __restrict__ A, const bf16_t* __restrict__ W,
                    const float* __restrict__ bias, void* __restrict__ dst,
                    int mode) {
  const int lane = threadIdx.x & 31;
  const int wave = threadIdx.x >> 5;
  const int w    = blockIdx.x * 4 + wave;          // 2048 waves total
  const int NB   = EMBED / 64;                     // 16 n-blocks
  const int m0   = (w / NB) * 32;
  const int n0   = (w % NB) * 64;

  const int ln = lane & 15;
  const int hi = (lane & 16) ? 1 : 0;

  const bf16_t* arow0 = A + (size_t)(m0 + ln) * EMBED + hi * 16;
  const bf16_t* arow1 = arow0 + (size_t)16 * EMBED;
  const bf16_t* wrow[4];
#pragma unroll
  for (int t = 0; t < 4; ++t)
    wrow[t] = W + (size_t)(n0 + t * 16 + ln) * EMBED + hi * 16;

  f32x8 acc[2][4];
#pragma unroll
  for (int mi = 0; mi < 2; ++mi)
#pragma unroll
    for (int t = 0; t < 4; ++t)
      acc[mi][t] = (f32x8){0.f,0.f,0.f,0.f,0.f,0.f,0.f,0.f};

#pragma unroll 2
  for (int ks = 0; ks < EMBED / 32; ++ks) {
    const bf16x16 a0 = concat8(*(const bf16x8*)(arow0 + ks * 32),
                               *(const bf16x8*)(arow0 + ks * 32 + 8));
    const bf16x16 a1 = concat8(*(const bf16x8*)(arow1 + ks * 32),
                               *(const bf16x8*)(arow1 + ks * 32 + 8));
#pragma unroll
    for (int t = 0; t < 4; ++t) {
      const bf16x16 bf_ = frag_b(wrow[t] + ks * 32);
      acc[0][t] = wmma_bf16(a0, bf_, acc[0][t]);
      acc[1][t] = wmma_bf16(a1, bf_, acc[1][t]);
    }
  }

#pragma unroll
  for (int t = 0; t < 4; ++t) {
    const int col = n0 + t * 16 + ln;
    const float bv = bias[col];
    const int hh = col >> 6;        // head
    const int dd = col & 63;        // dim in head
#pragma unroll
    for (int mi = 0; mi < 2; ++mi) {
#pragma unroll
      for (int j = 0; j < 8; ++j) {
        const int row = m0 + mi * 16 + j + hi * 8;    // token index
        const int bb  = row >> 11;                    // /SEQ
        const int ss  = row & (SEQ - 1);
        const float v = acc[mi][t][j] + bv;
        if (mode == 0) {
          ((bf16_t*)dst)[(((size_t)(bb * HEADS + hh) * SEQ + ss) * HDIM) + dd] = (bf16_t)v;
        } else if (mode == 1) {
          ((bf16_t*)dst)[(((size_t)(bb * HEADS + hh) * HDIM + dd) * SEQ) + ss] = (bf16_t)v;
        } else {
          ((float*)dst)[(size_t)row * EMBED + col] = v;
        }
      }
    }
  }
}

// ---------------------------------------------------------------------------
// Flash attention: one wave owns 16 query rows of one (b,h); streams keys in
// blocks of 32 with online softmax; ctx accumulated in 4 f32x8 fragments.
__global__ __launch_bounds__(128)
void attn_wmma(const bf16_t* __restrict__ Qh, const bf16_t* __restrict__ Kh,
               const bf16_t* __restrict__ Vt, bf16_t* __restrict__ Ctx) {
  __shared__ __align__(16) bf16_t pbuf[4][16][40];   // 16x32 P tile + pad

  const int lane = threadIdx.x & 31;
  const int wave = threadIdx.x >> 5;
  const int id   = blockIdx.x * 4 + wave;            // 4096 q-tiles
  const int qt   = id & (SEQ / 16 - 1);              // 0..127
  const int bh   = id >> 7;                          // 0..31
  const int q0   = qt * 16;

  const int ln = lane & 15;
  const int hi = (lane & 16) ? 1 : 0;

  const bf16_t* Qbase = Qh + ((size_t)bh * SEQ + q0) * HDIM;
  const bf16_t* Kbase = Kh + (size_t)bh * SEQ * HDIM;
  const bf16_t* Vbase = Vt + (size_t)bh * HDIM * SEQ;

  // Q A-fragments for the two 32-wide d slices (loaded once).
  const bf16x16 qa0 = frag_a(Qbase, HDIM, lane);
  const bf16x16 qa1 = frag_a(Qbase + 32, HDIM, lane);

  f32x8 acc0 = {0.f,0.f,0.f,0.f,0.f,0.f,0.f,0.f};
  f32x8 acc1 = acc0, acc2 = acc0, acc3 = acc0;
  float mrun[8], lrun[8];
#pragma unroll
  for (int j = 0; j < 8; ++j) { mrun[j] = -1e30f; lrun[j] = 0.f; }

  for (int kb = 0; kb < SEQ; kb += 32) {
    // ---- scores: two 16x16 fragments over key cols [kb..kb+15],[kb+16..31]
    const bf16_t* k0p = Kbase + (size_t)(kb + ln) * HDIM + hi * 16;
    const bf16_t* k1p = Kbase + (size_t)(kb + 16 + ln) * HDIM + hi * 16;
    const bf16_t* vb  = Vbase + (size_t)ln * SEQ + kb + hi * 16;
    __builtin_prefetch(k0p + 32 * HDIM, 0, 1);       // next key block
    __builtin_prefetch(vb + 32, 0, 1);               // next value block

    f32x8 s0 = {0.f,0.f,0.f,0.f,0.f,0.f,0.f,0.f};
    f32x8 s1 = s0;
    s0 = wmma_bf16(qa0, frag_b(k0p), s0);
    s0 = wmma_bf16(qa1, frag_b(k0p + 32), s0);
    s1 = wmma_bf16(qa0, frag_b(k1p), s1);
    s1 = wmma_bf16(qa1, frag_b(k1p + 32), s1);

    // ---- online softmax per row (row j lives in VGPR j within a 16-lane half)
#pragma unroll
    for (int j = 0; j < 8; ++j) {
      const float v0 = s0[j] * 0.125f;               // 1/sqrt(64)
      const float v1 = s1[j] * 0.125f;
      float mb = fmaxf(v0, v1);
      mb = fmaxf(mb, __shfl_xor(mb, 1, 16));
      mb = fmaxf(mb, __shfl_xor(mb, 2, 16));
      mb = fmaxf(mb, __shfl_xor(mb, 4, 16));
      mb = fmaxf(mb, __shfl_xor(mb, 8, 16));
      const float mn   = fmaxf(mrun[j], mb);
      const float corr = __expf(mrun[j] - mn);
      const float e0   = __expf(v0 - mn);
      const float e1   = __expf(v1 - mn);
      float rs = e0 + e1;
      rs += __shfl_xor(rs, 1, 16);
      rs += __shfl_xor(rs, 2, 16);
      rs += __shfl_xor(rs, 4, 16);
      rs += __shfl_xor(rs, 8, 16);
      lrun[j] = lrun[j] * corr + rs;
      mrun[j] = mn;
      acc0[j] *= corr; acc1[j] *= corr; acc2[j] *= corr; acc3[j] *= corr;
      // stash P in C/D layout: row = j + 8*hi, col = ln (+16 for second frag)
      pbuf[wave][j + hi * 8][ln]      = (bf16_t)e0;
      pbuf[wave][j + hi * 8][ln + 16] = (bf16_t)e1;
    }

    // per-wave LDS RAW: wait on the DS counter before re-reading (CDNA5 split
    // counters), then pull P back in A-fragment layout.
    asm volatile("s_wait_dscnt 0" ::: "memory");
    const bf16_t* pp = &pbuf[wave][ln][hi * 16];
    bf16x8 pa0 = *(const bf16x8*)(pp);
    bf16x8 pa1 = *(const bf16x8*)(pp + 8);
    const bf16x16 pf = concat8(pa0, pa1);

    // ---- ctx += P @ V  (V stored transposed [d, s]; contiguous in keys)
    acc0 = wmma_bf16(pf, frag_b(vb + 0 * 16 * SEQ), acc0);
    acc1 = wmma_bf16(pf, frag_b(vb + 1 * 16 * SEQ), acc1);
    acc2 = wmma_bf16(pf, frag_b(vb + 2 * 16 * SEQ), acc2);
    acc3 = wmma_bf16(pf, frag_b(vb + 3 * 16 * SEQ), acc3);
  }

  // ---- normalize and write ctx as bf16 in [b, s, h*d] for the output GEMM
  const int bb = bh >> 4;
  const int hh = bh & 15;
#pragma unroll
  for (int j = 0; j < 8; ++j) {
    const float inv = 1.0f / lrun[j];
    const int row = q0 + j + hi * 8;
    bf16_t* dst = Ctx + ((size_t)(bb * SEQ + row) * EMBED) + hh * HDIM;
    dst[0 * 16 + ln] = (bf16_t)(acc0[j] * inv);
    dst[1 * 16 + ln] = (bf16_t)(acc1[j] * inv);
    dst[2 * 16 + ln] = (bf16_t)(acc2[j] * inv);
    dst[3 * 16 + ln] = (bf16_t)(acc3[j] * inv);
  }
}

// ---------------------------------------------------------------------------
extern "C" void kernel_launch(void* const* d_in, const int* in_sizes, int n_in,
                              void* d_out, int out_size, void* d_ws, size_t ws_size,
                              hipStream_t stream) {
  (void)in_sizes; (void)n_in; (void)out_size; (void)ws_size;

  const float* query  = (const float*)d_in[0];
  const float* key_in = (const float*)d_in[1];
  const float* value  = (const float*)d_in[2];
  const float* Wq = (const float*)d_in[3];
  const float* bq = (const float*)d_in[4];
  const float* Wk = (const float*)d_in[5];
  const float* bk = (const float*)d_in[6];
  const float* Wv = (const float*)d_in[7];
  const float* bv = (const float*)d_in[8];
  const float* Wo = (const float*)d_in[9];
  const float* bo = (const float*)d_in[10];

  const size_t ACT_N = (size_t)TOKENS * EMBED;   // 4M elements
  const size_t WT_N  = (size_t)EMBED * EMBED;    // 1M elements

  size_t off = 0;
  auto alloc = [&](size_t bytes) {
    void* p = (char*)d_ws + off;
    off += (bytes + 255) & ~(size_t)255;
    return p;
  };
  bf16_t* Xq  = (bf16_t*)alloc(ACT_N * 2);
  bf16_t* Xk  = (bf16_t*)alloc(ACT_N * 2);
  bf16_t* Xv  = (bf16_t*)alloc(ACT_N * 2);
  bf16_t* Wqb = (bf16_t*)alloc(WT_N * 2);
  bf16_t* Wkb = (bf16_t*)alloc(WT_N * 2);
  bf16_t* Wvb = (bf16_t*)alloc(WT_N * 2);
  bf16_t* Wob = (bf16_t*)alloc(WT_N * 2);
  bf16_t* Qh  = (bf16_t*)alloc(ACT_N * 2);       // [b,h,s,d]
  bf16_t* Khd = (bf16_t*)alloc(ACT_N * 2);       // [b,h,s,d]
  bf16_t* Vt  = (bf16_t*)alloc(ACT_N * 2);       // [b,h,d,s]
  bf16_t* Ctx = (bf16_t*)alloc(ACT_N * 2);       // [b,s,e]

  const int CVT_B = 256;
  const int ACT4 = (int)(ACT_N / 4), WT4 = (int)(WT_N / 4);
  cvt_f32_to_bf16<<<(ACT4 + CVT_B - 1) / CVT_B, CVT_B, 0, stream>>>(query,  Xq, ACT4);
  cvt_f32_to_bf16<<<(ACT4 + CVT_B - 1) / CVT_B, CVT_B, 0, stream>>>(key_in, Xk, ACT4);
  cvt_f32_to_bf16<<<(ACT4 + CVT_B - 1) / CVT_B, CVT_B, 0, stream>>>(value,  Xv, ACT4);
  cvt_f32_to_bf16<<<(WT4 + CVT_B - 1) / CVT_B, CVT_B, 0, stream>>>(Wq, Wqb, WT4);
  cvt_f32_to_bf16<<<(WT4 + CVT_B - 1) / CVT_B, CVT_B, 0, stream>>>(Wk, Wkb, WT4);
  cvt_f32_to_bf16<<<(WT4 + CVT_B - 1) / CVT_B, CVT_B, 0, stream>>>(Wv, Wvb, WT4);
  cvt_f32_to_bf16<<<(WT4 + CVT_B - 1) / CVT_B, CVT_B, 0, stream>>>(Wo, Wob, WT4);

  // 2048 waves (32x64 tile each), 4 waves (128 threads) per block
  const int GEMM_BLOCKS = (TOKENS / 32) * (EMBED / 64) / 4;   // 512
  gemm_bf16_wmma<<<GEMM_BLOCKS, 128, 0, stream>>>(Xq, Wqb, bq, Qh, 0);
  gemm_bf16_wmma<<<GEMM_BLOCKS, 128, 0, stream>>>(Xk, Wkb, bk, Khd, 0);
  gemm_bf16_wmma<<<GEMM_BLOCKS, 128, 0, stream>>>(Xv, Wvb, bv, Vt, 1);

  // 4096 q-tiles, 4 waves per block
  attn_wmma<<<BATCH * HEADS * (SEQ / 16) / 4, 128, 0, stream>>>(Qh, Khd, Vt, Ctx);

  gemm_bf16_wmma<<<GEMM_BLOCKS, 128, 0, stream>>>(Ctx, Wob, bo, d_out, 2);
}
